// NormalizationModelofAttention_14980845928448
// MI455X (gfx1250) — matrix operand order repaired and checked by reading, not compiled, not run on hardware
//
#include <hip/hip_runtime.h>

typedef __attribute__((ext_vector_type(16))) _Float16 v16h;
typedef __attribute__((ext_vector_type(8)))  _Float16 v8h;
typedef __attribute__((ext_vector_type(8)))  float    v8f;

#define GRID_N 64
#define NK     4096      // neurons / spatial positions
#define NT     256       // time frames
#define ECCF   10.0f

#define BK 64            // k rows per block
#define BA 32            // a cols per block
#define KK 64            // reduction chunk = one grid row i
#define LSTRIDE 72       // LDS row stride in halves (144B, 16B aligned)

__device__ __forceinline__ float coordv(int c) {
    return -ECCF + (2.0f * ECCF / (float)(GRID_N - 1)) * (float)c;
}

// ---- stim f32 -> f16 ----
__global__ void cvt_stim_kernel(const float* __restrict__ s,
                                _Float16* __restrict__ d, int n) {
    int i = blockIdx.x * blockDim.x + threadIdx.x;
    if (i < n) d[i] = (_Float16)s[i];
}

// ---- gx/gy normalized Gaussian tables (3 fields x 2 axes x 4096 k x 64) + attfield ----
__global__ void build_tables_kernel(const float* __restrict__ ssfP,
                                    const float* __restrict__ attSigP,
                                    const float* __restrict__ gainP,
                                    const float* __restrict__ surFacP,
                                    const float* __restrict__ sumFacP,
                                    float* __restrict__ gtab,
                                    float* __restrict__ attf) {
    int tid = blockIdx.x * blockDim.x + threadIdx.x;
    if (tid >= 3 * 2 * NK) return;
    int f    = tid >> 13;         // 0..2
    int rem  = tid & 8191;
    int axis = rem >> 12;         // 0 = x, 1 = y
    int k    = rem & (NK - 1);

    float cx = coordv(k & 63);    // RF_X[k]
    float cy = coordv(k >> 6);    // RF_Y[k]
    float fac = (f == 0) ? 1.0f : ((f == 1) ? surFacP[0] : sumFacP[0]);
    float sig = (0.07f + ssfP[0] * sqrtf(cx * cx + cy * cy)) * fac;
    float inv2s2 = 1.0f / (2.0f * sig * sig);
    float center = (axis == 0) ? cx : cy;

    float sum = 0.0f;
    for (int c = 0; c < GRID_N; ++c) {
        float d = coordv(c) - center;
        sum += __expf(-d * d * inv2s2);
    }
    float inv = 1.0f / sum;
    float* dst = gtab + ((size_t)(f * 2 + axis) * NK + (size_t)k) * GRID_N;
    for (int c = 0; c < GRID_N; ++c) {
        float d = coordv(c) - center;
        dst[c] = __expf(-d * d * inv2s2) * inv;
    }

    if (tid < NK) {  // f==0, axis==0 lanes also produce attfield
        float sa = attSigP[0];
        float g  = gainP[0];
        float dx = cx - 3.0f;     // ATT_CTR = (3, 0)
        attf[k] = g * __expf(-(dx * dx + cy * cy) / (2.0f * sa * sa)) + 1.0f;
    }
}

// ---- fused on-the-fly-field GEMM:  D[k,a] = sum_m gy[k][i_m]*gx[k][j_m] * B[m,a] ----
// mode 0: numerator = D * attf[k]           -> outMain (f32), outH (f16)
// mode 1: surround  = D                     -> outMain; pn = num/(D+0.5) -> outAux, outH
// mode 2: voxel     = D                     -> outMain
__global__ __launch_bounds__(128)
void fused_field_gemm(const float* __restrict__ gtab, int field,
                      const _Float16* __restrict__ B,      // [NK][NT] f16
                      const float* __restrict__ attf,
                      const float* __restrict__ numerIn,   // mode 1 only
                      float* __restrict__ outMain,
                      float* __restrict__ outAux,
                      _Float16* __restrict__ outH,
                      int mode) {
    __shared__ _Float16 Ash[BK * LSTRIDE];
    __shared__ _Float16 Bsh[BA * LSTRIDE];

    const int tid  = threadIdx.x;
    const int wave = tid >> 5;
    const int lane = tid & 31;
    const int M    = lane & 15;
    const int h    = lane >> 4;
    const int kBlock = blockIdx.x * BK;
    const int aBlock = blockIdx.y * BA;

    const float* gx = gtab + (size_t)(field * 2 + 0) * NK * GRID_N;
    const float* gy = gtab + (size_t)(field * 2 + 1) * NK * GRID_N;

    v8f acc0 = {};
    v8f acc1 = {};

    for (int it = 0; it < NK / KK; ++it) {
        __syncthreads();
        // Build A tile: Ash[r][c] = gy[kBlock+r][it] * gx[kBlock+r][c], c=0..63
        {
            int r  = tid >> 1;
            int hh = tid & 1;
            int k  = kBlock + r;
            float gyv = gy[(size_t)k * GRID_N + it];
            const float* gxr = gx + (size_t)k * GRID_N + hh * 32;
            _Float16* dst = Ash + r * LSTRIDE + hh * 32;
            #pragma unroll
            for (int c = 0; c < 32; ++c)
                dst[c] = (_Float16)(gyv * gxr[c]);
        }
        // Build transposed B tile: Bsh[a'][c] = B[(it*64+c)*NT + aBlock + a']
        {
            #pragma unroll
            for (int e = 0; e < 16; ++e) {
                int idx = tid + 128 * e;
                int ap  = idx & 31;
                int c   = idx >> 5;
                Bsh[ap * LSTRIDE + c] = B[(size_t)(it * KK + c) * NT + aBlock + ap];
            }
        }
        __syncthreads();

        // Two 16x16x32 WMMA steps per chunk, two N tiles per wave
        #pragma unroll
        for (int q = 0; q < 2; ++q) {
            // A fragment (16x32 f16): lane M=L%16, K halves {h*8+0..7, 16+h*8+0..7}
            const _Float16* arow = Ash + (wave * 16 + M) * LSTRIDE + q * 32 + h * 8;
            v8h alo = *(const v8h*)(arow);
            v8h ahi = *(const v8h*)(arow + 16);
            v16h afrag = __builtin_shufflevector(alo, ahi,
                0,1,2,3,4,5,6,7,8,9,10,11,12,13,14,15);
            #pragma unroll
            for (int n = 0; n < 2; ++n) {
                // B fragment (32x16 f16): lane column N=L&15, K halves {h*8+0..7, 16+h*8+0..7}
                const _Float16* brow = Bsh + (n * 16 + M) * LSTRIDE + q * 32 + h * 8;
                v8h blo = *(const v8h*)(brow);
                v8h bhi = *(const v8h*)(brow + 16);
                v16h bfrag = __builtin_shufflevector(blo, bhi,
                    0,1,2,3,4,5,6,7,8,9,10,11,12,13,14,15);
                if (n == 0)
                    acc0 = __builtin_amdgcn_wmma_f32_16x16x32_f16(
                        false, afrag, false, bfrag, (short)0, acc0, false, false);
                else
                    acc1 = __builtin_amdgcn_wmma_f32_16x16x32_f16(
                        false, afrag, false, bfrag, (short)0, acc1, false, false);
            }
        }
    }

    // Epilogue. C/D layout: VGPR v -> M = v + h*8 ; N = lane&15
    #pragma unroll
    for (int n = 0; n < 2; ++n) {
        v8f acc = (n == 0) ? acc0 : acc1;
        int a = aBlock + n * 16 + M;
        #pragma unroll
        for (int v = 0; v < 8; ++v) {
            int k = kBlock + wave * 16 + v + h * 8;
            size_t o = (size_t)k * NT + (size_t)a;
            float val = acc[v];
            if (mode == 0) {
                float num = val * attf[k];
                outMain[o] = num;
                outH[o] = (_Float16)num;
            } else if (mode == 1) {
                outMain[o] = val;                       // surroundresponse
                float pn = numerIn[o] / (val + 0.5f);   // NORM_SIGMA
                outAux[o] = pn;                         // predicted_neural
                outH[o] = (_Float16)pn;
            } else {
                outMain[o] = val;                       // predicted_voxel
            }
        }
    }
}

extern "C" void kernel_launch(void* const* d_in, const int* in_sizes, int n_in,
                              void* d_out, int out_size, void* d_ws, size_t ws_size,
                              hipStream_t stream) {
    const float* stim   = (const float*)d_in[0];
    const float* ssf    = (const float*)d_in[1];
    const float* attSig = (const float*)d_in[2];
    const float* gain   = (const float*)d_in[3];
    const float* surFac = (const float*)d_in[4];
    const float* sumFac = (const float*)d_in[5];

    float* out = (float*)d_out;
    char*  ws  = (char*)d_ws;

    float*    gtab    = (float*)(ws);                    // 3*2*4096*64 f32 = 6291456 B
    float*    attf    = (float*)(ws + 6291456);          // 4096 f32      =   16384 B
    _Float16* stim_h  = (_Float16*)(ws + 6307840);       // 4096*256 f16  = 2097152 B
    _Float16* numer_h = (_Float16*)(ws + 8404992);       // 2097152 B
    _Float16* pneur_h = (_Float16*)(ws + 10502144);      // 2097152 B  (total ~12.6 MB)

    const size_t O = (size_t)NK * NT;
    float* out_num = out;
    float* out_sur = out + O;
    float* out_pn  = out + 2 * O;
    float* out_pv  = out + 3 * O;

    cvt_stim_kernel<<<(NK * NT + 255) / 256, 256, 0, stream>>>(stim, stim_h, NK * NT);
    build_tables_kernel<<<(3 * 2 * NK + 127) / 128, 128, 0, stream>>>(
        ssf, attSig, gain, surFac, sumFac, gtab, attf);

    dim3 g(NK / BK, NT / BA), b(128);
    // GEMM1: stimulus_drive -> numerator (apply attfield in epilogue)
    fused_field_gemm<<<g, b, 0, stream>>>(gtab, 0, stim_h, attf,
                                          nullptr, out_num, nullptr, numer_h, 0);
    // GEMM2: surroundresponse + predicted_neural
    fused_field_gemm<<<g, b, 0, stream>>>(gtab, 1, numer_h, attf,
                                          out_num, out_sur, out_pn, pneur_h, 1);
    // GEMM3: predicted_voxel
    fused_field_gemm<<<g, b, 0, stream>>>(gtab, 2, pneur_h, attf,
                                          nullptr, out_pv, nullptr, nullptr, 2);
}
